// CombinedModel_34050500722843
// MI455X (gfx1250) — compile-verified, matrix operand order checked by reading, not run on hardware
//
#include <hip/hip_runtime.h>
#include <hip/hip_bf16.h>

// ---------------------------------------------------------------------------
// CDNA5 / gfx1250 implementation.
// Matrix work: v_wmma_f32_16x16x32_f16 fed from double-buffered LDS tiles
// filled with global_load_async_to_lds_b128 (ASYNCcnt) when available.
// SAGE mean-aggregation: global f32 atomics.
// ---------------------------------------------------------------------------

typedef _Float16 v16h __attribute__((ext_vector_type(16)));
typedef _Float16 v8h  __attribute__((ext_vector_type(8)));
typedef float    v8f  __attribute__((ext_vector_type(8)));
typedef int      i4v  __attribute__((__vector_size__(16)));   // 128-bit payload type

#define AS1 __attribute__((address_space(1)))
#define AS3 __attribute__((address_space(3)))
#if __has_builtin(__builtin_amdgcn_global_load_async_to_lds_b128) && \
    __has_builtin(__builtin_amdgcn_s_wait_asynccnt)
#define HAVE_ASYNC_LDS 1
#else
#define HAVE_ASYNC_LDS 0
#endif

// Problem constants (fixed by the reference).
constexpr int Nn  = 50000;
constexpr int Ee  = 800000;
constexpr int IND = 128;
constexpr int Hd  = 256;
constexpr int Bb  = 4;
constexpr int Td  = 512;
constexpr int BT  = Bb * Td;                 // 2048
constexpr int NPAD = ((Nn + 63) / 64) * 64;  // 50048

// LDS tile geometry: rows padded to 40 halves (80 B) to spread LDS banks.
constexpr int LROW = 40;

// --------------------------- WMMA fragment helpers -------------------------
// 16-bit A operand 16x32 (B operand 32x16 has the same load pattern for
// row-major [rows, K] storage): lanes 0-15 -> row=lane, K chunks [0..7] and
// [16..23]; lanes 16-31 -> row=lane-16, K chunks [8..15] and [24..31].
__device__ inline v16h load_frag(const _Float16* p) {
  union { v16h v; struct { v8h lo; v8h hi; } s; } u;
  u.s.lo = *(const v8h*)(p);        // 16B -> ds_load_b128 / global_load_b128
  u.s.hi = *(const v8h*)(p + 16);
  return u.v;
}

__device__ inline void async_cp_b128(const _Float16* g, _Float16* l) {
#if HAVE_ASYNC_LDS
  __builtin_amdgcn_global_load_async_to_lds_b128((AS1 i4v*)(g), (AS3 i4v*)(l), 0, 0);
#else
  *(v8h*)l = *(const v8h*)g;
#endif
}

// --------------------------- tiled WMMA GEMM -------------------------------
// D[M, Nc] = act( A1[M,K1] @ B1[Nc,K1]^T (+ A2[M,K2] @ B2[Nc,K2]^T) + bias )
// Block = 256 threads = 8 waves. Tile = 128(M) x 64(N); each wave owns a
// 16x64 strip (4 accumulators). A/B K-chunks staged in LDS, double buffered,
// async-copied one chunk ahead of the WMMAs.
__global__ __launch_bounds__(256)
void wmma_gemm(const _Float16* __restrict__ A1, const _Float16* __restrict__ B1, int K1,
               const _Float16* __restrict__ A2, const _Float16* __restrict__ B2, int K2,
               const float* __restrict__ bias,
               float* __restrict__ outF, _Float16* __restrict__ outH,
               int M, int Nstore, int ldd, int relu)
{
  __shared__ _Float16 lA[2][128 * LROW];   // 2 x 10240 B
  __shared__ _Float16 lB[2][64 * LROW];    // 2 x  5120 B

  const int tid  = threadIdx.x;
  const int wave = tid >> 5;
  const int lane = tid & 31;
  const int m0 = blockIdx.y * 128;
  const int n0 = blockIdx.x * 64;
  const int mw = m0 + wave * 16;           // this wave's M strip (may exceed M)

  const int nsteps1 = K1 >> 5;
  const int nsteps2 = A2 ? (K2 >> 5) : 0;
  const int nsteps  = nsteps1 + nsteps2;

  // Stage one 32-wide K chunk of A (128 rows) + B (64 rows) into LDS buffer.
  // Each thread issues exactly 3 x b128 copies (2 for A, 1 for B).
  auto stage = [&](int buf, int s) {
    const _Float16* A; const _Float16* B; int ld, k;
    if (s < nsteps1) { A = A1; B = B1; ld = K1; k = s << 5; }
    else             { A = A2; B = B2; ld = K2; k = (s - nsteps1) << 5; }
#pragma unroll
    for (int i = 0; i < 2; ++i) {
      int id = tid + i * 256;              // 512 chunks: 128 rows x 4
      int r  = id >> 2;
      int c8 = (id & 3) << 3;
      int gr = min(m0 + r, M - 1);         // clamp tail rows (M % 16 == 0)
      async_cp_b128(A + (size_t)gr * ld + k + c8, &lA[buf][r * LROW + c8]);
    }
    {
      int r  = tid >> 2;                   // 256 chunks: 64 rows x 4
      int c8 = (tid & 3) << 3;
      async_cp_b128(B + (size_t)(n0 + r) * ld + k + c8, &lB[buf][r * LROW + c8]);
    }
  };

  v8f c[4] = {};
  const int row  = lane & 15;
  const int kofs = (lane < 16) ? 0 : 8;

  stage(0, 0);
  for (int s = 0; s < nsteps; ++s) {
    const int buf = s & 1;
    if (s + 1 < nsteps) {
      stage(buf ^ 1, s + 1);               // prefetch next chunk while computing
#if HAVE_ASYNC_LDS
      __builtin_amdgcn_s_wait_asynccnt(3); // retire current chunk's 3 copies
#endif
    } else {
#if HAVE_ASYNC_LDS
      __builtin_amdgcn_s_wait_asynccnt(0);
#endif
    }
    __syncthreads();                       // all waves' copies visible

    v16h a = load_frag(&lA[buf][(wave * 16 + row) * LROW + kofs]);
    v16h b0 = load_frag(&lB[buf][(0 * 16 + row) * LROW + kofs]);
    v16h b1 = load_frag(&lB[buf][(1 * 16 + row) * LROW + kofs]);
    v16h b2 = load_frag(&lB[buf][(2 * 16 + row) * LROW + kofs]);
    v16h b3 = load_frag(&lB[buf][(3 * 16 + row) * LROW + kofs]);
    c[0] = __builtin_amdgcn_wmma_f32_16x16x32_f16(false, a, false, b0, (short)0, c[0], false, false);
    c[1] = __builtin_amdgcn_wmma_f32_16x16x32_f16(false, a, false, b1, (short)0, c[1], false, false);
    c[2] = __builtin_amdgcn_wmma_f32_16x16x32_f16(false, a, false, b2, (short)0, c[2], false, false);
    c[3] = __builtin_amdgcn_wmma_f32_16x16x32_f16(false, a, false, b3, (short)0, c[3], false, false);

    __syncthreads();                       // buffer may be overwritten next iter
  }

  if (mw >= M) return;                     // tail waves computed on clamped rows

  // C/D 16x16 f32 layout: lanes 0-15 -> n=lane, M=vgpr; lanes 16-31 -> n=lane-16, M=8+vgpr
  const int nLane = lane & 15;
  const int mofs  = (lane < 16) ? 0 : 8;
#pragma unroll
  for (int j = 0; j < 4; ++j) {
    int n = n0 + j * 16 + nLane;
    if (n >= Nstore) continue;
    float bv = bias ? bias[n] : 0.f;
#pragma unroll
    for (int r = 0; r < 8; ++r) {
      float v = c[j][r] + bv;
      if (relu) v = fmaxf(v, 0.f);
      size_t idx = (size_t)(mw + mofs + r) * (size_t)ldd + (size_t)n;
      if (outF) outF[idx] = v;
      if (outH) outH[idx] = (_Float16)v;
    }
  }
}

// --------------------------- SAGE scatter-add ------------------------------
__global__ void sage_scatter(const float* __restrict__ x, const int* __restrict__ src,
                             const int* __restrict__ dst, float* __restrict__ agg,
                             float* __restrict__ cnt, int E, int D, int tpeShift)
{
  long long gid = (long long)blockIdx.x * blockDim.x + threadIdx.x;
  int e = (int)(gid >> tpeShift);
  if (e >= E) return;
  int c4 = ((int)gid & ((1 << tpeShift) - 1)) << 2;
  int s = src[e], d = dst[e];
  float4 v = *(const float4*)(x + (size_t)s * D + c4);
  float* ap = agg + (size_t)d * D + c4;
  atomicAdd(ap + 0, v.x);
  atomicAdd(ap + 1, v.y);
  atomicAdd(ap + 2, v.z);
  atomicAdd(ap + 3, v.w);
  if (c4 == 0) atomicAdd(cnt + d, 1.0f);
}

// --------------------------- small pointwise kernels -----------------------
__global__ void cvt_f16(const float* __restrict__ in, _Float16* __restrict__ out, int n) {
  int i = blockIdx.x * blockDim.x + threadIdx.x;
  if (i < n) out[i] = (_Float16)in[i];
}

__global__ void mean_to_h(const float* __restrict__ agg, const float* __restrict__ cnt,
                          _Float16* __restrict__ o, int total, int dShift) {
  int i = blockIdx.x * blockDim.x + threadIdx.x;
  if (i >= total) return;
  int r = i >> dShift;
  o[i] = (_Float16)(agg[i] / fmaxf(cnt[r], 1.0f));
}

__global__ void add_vec(const float* __restrict__ a, const float* __restrict__ b,
                        float* __restrict__ o, int n) {
  int i = blockIdx.x * blockDim.x + threadIdx.x;
  if (i < n) o[i] = a[i] + b[i];
}

// Per-channel sum / sum-of-squares over rows (block = 256 ch, 128 rows/block).
__global__ __launch_bounds__(256)
void bn_stats(const float* __restrict__ x, float* __restrict__ sums, int N) {
  int ch = threadIdx.x;
  int r0 = blockIdx.x * 128;
  int r1 = min(r0 + 128, N);
  float s = 0.f, s2 = 0.f;
  for (int r = r0; r < r1; ++r) {
    float v = x[(size_t)r * Hd + ch];
    s += v; s2 += v * v;
  }
  atomicAdd(&sums[ch], s);
  atomicAdd(&sums[Hd + ch], s2);
}

// x_out = relu( BN(pre)*g + b + skip ); optional f32 and f16 outputs.
__global__ void bn_skip_relu(const float* __restrict__ pre, const float* __restrict__ skip,
                             const float* __restrict__ sums,
                             const float* __restrict__ g, const float* __restrict__ b,
                             float* __restrict__ outF, _Float16* __restrict__ outH, int N) {
  int i = blockIdx.x * blockDim.x + threadIdx.x;
  if (i >= N * Hd) return;
  int ch = i & (Hd - 1);
  float mean = sums[ch] / (float)N;
  float var  = sums[Hd + ch] / (float)N - mean * mean;
  float v = (pre[i] - mean) * rsqrtf(var + 1e-5f) * g[ch] + b[ch] + skip[i];
  v = fmaxf(v, 0.f);
  if (outF) outF[i] = v;
  outH[i] = (_Float16)v;
}

// --------------------------- BiLSTM recurrence -----------------------------
__device__ inline float sigmoidf(float x) { return 1.0f / (1.0f + __expf(-x)); }

// grid = (B, 2 dirs); block = 512 threads (one per gate output).
__global__ __launch_bounds__(512)
void lstm_scan(const float* __restrict__ xs_f, const float* __restrict__ xs_b,
               const float* __restrict__ whh_f, const float* __restrict__ whh_b,
               _Float16* __restrict__ outH)
{
  const int b   = blockIdx.x;
  const int dir = blockIdx.y;
  const float* xs  = dir ? xs_b  : xs_f;   // [BT, 512], row = b*T + t
  const float* whh = dir ? whh_b : whh_f;  // [512, 128] f32
  __shared__ float h[128], c[128], gates[512];
  const int tid = threadIdx.x;
  if (tid < 128) { h[tid] = 0.f; c[tid] = 0.f; }
  __syncthreads();
  for (int s = 0; s < Td; ++s) {
    int t = dir ? (Td - 1 - s) : s;
    float acc = xs[((size_t)b * Td + t) * 512 + tid];
    const float* wrow = whh + (size_t)tid * 128;
#pragma unroll 8
    for (int k = 0; k < 128; ++k) acc += wrow[k] * h[k];
    gates[tid] = acc;
    __syncthreads();
    if (tid < 128) {
      float ig = sigmoidf(gates[tid]);
      float fg = sigmoidf(gates[128 + tid]);
      float gg = tanhf(gates[256 + tid]);
      float og = sigmoidf(gates[384 + tid]);
      float cn = fg * c[tid] + ig * gg;
      float hn = og * tanhf(cn);
      c[tid] = cn; h[tid] = hn;
      outH[((size_t)b * Td + t) * Hd + dir * 128 + tid] = (_Float16)hn;
    }
    __syncthreads();
  }
}

// ---------------------------------------------------------------------------
extern "C" void kernel_launch(void* const* d_in, const int* in_sizes, int n_in,
                              void* d_out, int out_size, void* d_ws, size_t ws_size,
                              hipStream_t stream) {
  (void)in_sizes; (void)n_in; (void)out_size; (void)ws_size;
  const float* trace = (const float*)d_in[0];
  const float* x     = (const float*)d_in[1];
  const int*   ei    = (const int*)d_in[2];
  const int*   src   = ei;
  const int*   dst   = ei + Ee;
  const float* c1wl = (const float*)d_in[3];  const float* c1bl = (const float*)d_in[4];
  const float* c1wr = (const float*)d_in[5];
  const float* c2wl = (const float*)d_in[6];  const float* c2bl = (const float*)d_in[7];
  const float* c2wr = (const float*)d_in[8];
  const float* s1w1 = (const float*)d_in[9];  const float* s1b1 = (const float*)d_in[10];
  const float* s1w2 = (const float*)d_in[11]; const float* s1b2 = (const float*)d_in[12];
  const float* s2w1 = (const float*)d_in[13]; const float* s2b1 = (const float*)d_in[14];
  const float* s2w2 = (const float*)d_in[15]; const float* s2b2 = (const float*)d_in[16];
  const float* bn1g = (const float*)d_in[17]; const float* bn1b = (const float*)d_in[18];
  const float* bn2g = (const float*)d_in[19]; const float* bn2b = (const float*)d_in[20];
  const float* wihf = (const float*)d_in[21]; const float* whhf = (const float*)d_in[22];
  const float* bihf = (const float*)d_in[23]; const float* bhhf = (const float*)d_in[24];
  const float* wihb = (const float*)d_in[25]; const float* whhb = (const float*)d_in[26];
  const float* bihb = (const float*)d_in[27]; const float* bhhb = (const float*)d_in[28];
  float* out = (float*)d_out;

  // ---- workspace layout (256B aligned slices) ----
  char* wsb = (char*)d_ws;
  size_t off = 0;
  auto take = [&](size_t bytes) -> void* {
    void* p = wsb + off;
    off = (off + bytes + 255) & ~(size_t)255;
    return p;
  };
  float*    bufA   = (float*)take((size_t)Nn * Hd * 4);   // agg (conv) then reused as pre-BN
  float*    cntB   = (float*)take((size_t)Nn * 4);
  float*    x1     = (float*)take((size_t)Nn * Hd * 4);
  float*    skip   = (float*)take((size_t)Nn * Hd * 4);
  _Float16* meanh  = (_Float16*)take((size_t)Nn * Hd * 2);
  _Float16* xh     = (_Float16*)take((size_t)Nn * IND * 2);
  _Float16* x1h    = (_Float16*)take((size_t)Nn * Hd * 2);
  _Float16* hidh   = (_Float16*)take((size_t)Nn * 128 * 2);
  _Float16* x2h    = (_Float16*)take((size_t)NPAD * Hd * 2);
  float*    bnsums = (float*)take(2 * Hd * 4);
  _Float16* c1wl_h = (_Float16*)take((size_t)Hd * IND * 2);
  _Float16* c1wr_h = (_Float16*)take((size_t)Hd * IND * 2);
  _Float16* c2wl_h = (_Float16*)take((size_t)Hd * Hd * 2);
  _Float16* c2wr_h = (_Float16*)take((size_t)Hd * Hd * 2);
  _Float16* s1w1_h = (_Float16*)take((size_t)128 * IND * 2);
  _Float16* s1w2_h = (_Float16*)take((size_t)Hd * 128 * 2);
  _Float16* s2w1_h = (_Float16*)take((size_t)128 * Hd * 2);
  _Float16* s2w2_h = (_Float16*)take((size_t)Hd * 128 * 2);
  _Float16* wihf_h = (_Float16*)take((size_t)512 * IND * 2);
  _Float16* wihb_h = (_Float16*)take((size_t)512 * IND * 2);
  _Float16* trc_h  = (_Float16*)take((size_t)BT * IND * 2);
  float*    cbias  = (float*)take((size_t)2 * 512 * 4);
  float*    xsf    = (float*)take((size_t)BT * 512 * 4);
  float*    xsb    = (float*)take((size_t)BT * 512 * 4);
  _Float16* lstm_h = (_Float16*)take((size_t)BT * Hd * 2);

  auto cvt = [&](const float* in, _Float16* o, int n) {
    cvt_f16<<<(n + 255) / 256, 256, 0, stream>>>(in, o, n);
  };
  auto gemm = [&](const _Float16* A1, const _Float16* B1, int K1,
                  const _Float16* A2, const _Float16* B2, int K2,
                  const float* bias, float* oF, _Float16* oH,
                  int M, int Ncols, int Nstore, int ldd, int relu) {
    dim3 g((Ncols + 63) / 64, (M + 127) / 128);
    wmma_gemm<<<g, 256, 0, stream>>>(A1, B1, K1, A2, B2, K2, bias, oF, oH,
                                     M, Nstore, ldd, relu);
  };
  auto memz = [&](void* p, size_t bytes) {
    (void)hipMemsetAsync(p, 0, bytes, stream);
  };

  // ---- one-time (per call) f16 conversions ----
  cvt(x, xh, Nn * IND);
  cvt(trace, trc_h, BT * IND);
  cvt(c1wl, c1wl_h, Hd * IND);  cvt(c1wr, c1wr_h, Hd * IND);
  cvt(c2wl, c2wl_h, Hd * Hd);   cvt(c2wr, c2wr_h, Hd * Hd);
  cvt(s1w1, s1w1_h, 128 * IND); cvt(s1w2, s1w2_h, Hd * 128);
  cvt(s2w1, s2w1_h, 128 * Hd);  cvt(s2w2, s2w2_h, Hd * 128);
  cvt(wihf, wihf_h, 512 * IND); cvt(wihb, wihb_h, 512 * IND);
  add_vec<<<2, 512, 0, stream>>>(bihf, bhhf, cbias, 512);
  add_vec<<<2, 512, 0, stream>>>(bihb, bhhb, cbias + 512, 512);

  // ================= SAGE layer 1 =================
  memz(bufA, (size_t)Nn * IND * 4);
  memz(cntB, (size_t)Nn * 4);
  {
    long long tot = (long long)Ee * (IND / 4);
    sage_scatter<<<(int)((tot + 255) / 256), 256, 0, stream>>>(
        x, src, dst, bufA, cntB, Ee, IND, 5);
  }
  mean_to_h<<<(Nn * IND + 255) / 256, 256, 0, stream>>>(bufA, cntB, meanh, Nn * IND, 7);
  // pre = mean@wl^T + x@wr^T + bl   (written into bufA, which is free now)
  gemm(meanh, c1wl_h, IND, xh, c1wr_h, IND, c1bl, bufA, nullptr, Nn, Hd, Hd, Hd, 0);
  // skip MLP: hid = relu(x@w1^T+b1) ; skip = hid@w2^T + b2
  gemm(xh, s1w1_h, IND, nullptr, nullptr, 0, s1b1, nullptr, hidh, Nn, 128, 128, 128, 1);
  gemm(hidh, s1w2_h, 128, nullptr, nullptr, 0, s1b2, skip, nullptr, Nn, Hd, Hd, Hd, 0);
  memz(bnsums, 2 * Hd * 4);
  bn_stats<<<(Nn + 127) / 128, 256, 0, stream>>>(bufA, bnsums, Nn);
  bn_skip_relu<<<(Nn * Hd + 255) / 256, 256, 0, stream>>>(
      bufA, skip, bnsums, bn1g, bn1b, x1, x1h, Nn);

  // ================= SAGE layer 2 =================
  memz(bufA, (size_t)Nn * Hd * 4);
  memz(cntB, (size_t)Nn * 4);
  {
    long long tot = (long long)Ee * (Hd / 4);
    sage_scatter<<<(int)((tot + 255) / 256), 256, 0, stream>>>(
        x1, src, dst, bufA, cntB, Ee, Hd, 6);
  }
  mean_to_h<<<(Nn * Hd + 255) / 256, 256, 0, stream>>>(bufA, cntB, meanh, Nn * Hd, 8);
  gemm(meanh, c2wl_h, Hd, x1h, c2wr_h, Hd, c2bl, bufA, nullptr, Nn, Hd, Hd, Hd, 0);
  gemm(x1h, s2w1_h, Hd, nullptr, nullptr, 0, s2b1, nullptr, hidh, Nn, 128, 128, 128, 1);
  gemm(hidh, s2w2_h, 128, nullptr, nullptr, 0, s2b2, skip, nullptr, Nn, Hd, Hd, Hd, 0);
  memz(bnsums, 2 * Hd * 4);
  bn_stats<<<(Nn + 127) / 128, 256, 0, stream>>>(bufA, bnsums, Nn);
  memz(x2h, (size_t)NPAD * Hd * 2);        // zero incl. N-pad rows
  bn_skip_relu<<<(Nn * Hd + 255) / 256, 256, 0, stream>>>(
      bufA, skip, bnsums, bn2g, bn2b, nullptr, x2h, Nn);

  // ================= BiLSTM =================
  // xs = trace @ wih^T + (bih + bhh), both directions.
  gemm(trc_h, wihf_h, IND, nullptr, nullptr, 0, cbias, xsf, nullptr, BT, 512, 512, 512, 0);
  gemm(trc_h, wihb_h, IND, nullptr, nullptr, 0, cbias + 512, xsb, nullptr, BT, 512, 512, 512, 0);
  lstm_scan<<<dim3(Bb, 2), 512, 0, stream>>>(xsf, xsb, whhf, whhb, lstm_h);

  // ================= final einsum: [BT,256] x [N,256]^T -> [BT,N] =========
  // x2h fits in L2 (25 MB << 192 MB); HBM traffic ~= read-once + 410 MB write.
  gemm(lstm_h, x2h, Hd, nullptr, nullptr, 0, nullptr, out, nullptr,
       BT, Nn, Nn, Nn, 0);
}
